// Exp_model_21595095564536
// MI455X (gfx1250) — compile-verified, matrix operand order checked by reading, not run on hardware
//
#include <hip/hip_runtime.h>

// -------------------- problem constants (match reference) --------------------
#define N_RECC   2048
#define N_INP    256
#define BATCH    64
#define TSTEPS   500
#define KTOT     (N_INP + N_RECC)      // 2304: fused K axis [W_in ; W_rec]
#define DECAYF   0.9512294245007140f   // exp(-1/20)
#define THRF     1.0f

// -------------------- tiling --------------------
#define KTILE_F32   64
#define STR_F32     68                 // floats; 16B-aligned rows, conflict-free frags
#define KTILE_BF    128
#define STR_BF      136                // ushorts; 272B rows = 17*16B -> aligned + conflict-free

typedef float  v2f   __attribute__((ext_vector_type(2)));
typedef float  v8f   __attribute__((ext_vector_type(8)));
typedef __bf16 v8bf  __attribute__((ext_vector_type(8)));
typedef __bf16 v16bf __attribute__((ext_vector_type(16)));
typedef int    v4i_  __attribute__((vector_size(16)));   // matches async-LDS builtin param

// -------------------- bf16 bit helpers (RNE, via integer ops) --------------------
__device__ __forceinline__ unsigned short f32_to_bf16_rne(float f) {
    unsigned int u = __float_as_uint(f);
    unsigned int lsb = (u >> 16) & 1u;
    u += 0x7fffu + lsb;
    return (unsigned short)(u >> 16);
}
__device__ __forceinline__ float bf16_to_f32(unsigned short h) {
    return __uint_as_float((unsigned int)h << 16);
}
__device__ __forceinline__ unsigned int pack_bf16(float a, float b) {
    return (unsigned int)f32_to_bf16_rne(a) | ((unsigned int)f32_to_bf16_rne(b) << 16);
}

// -------------------- async global->LDS (CDNA5 ASYNCcnt path) --------------------
#if defined(__gfx1250__) && __has_builtin(__builtin_amdgcn_global_load_async_to_lds_b128) && \
    __has_builtin(__builtin_amdgcn_s_wait_asynccnt)
#define HAVE_ASYNC_LDS 1
#else
#define HAVE_ASYNC_LDS 0
#endif

__device__ __forceinline__ void async_cp16(const unsigned short* g, unsigned short* l) {
#if HAVE_ASYNC_LDS
    // generic->AS1 / AS3 via integer roundtrip (LDS generic ptr: low 32 bits = LDS offset)
    __builtin_amdgcn_global_load_async_to_lds_b128(
        (__attribute__((address_space(1))) v4i_*)(unsigned long long)g,
        (__attribute__((address_space(3))) v4i_*)(unsigned int)(unsigned long long)l,
        0, 0);
#else
    *(uint4*)l = *(const uint4*)g;
#endif
}
// Wait until at most the newest chunk's 12 copies remain in flight
// (async loads complete in order -> the previous chunk has fully landed).
__device__ __forceinline__ void async_wait_prev() {
#if HAVE_ASYNC_LDS
    __builtin_amdgcn_s_wait_asynccnt(12);
#endif
}
__device__ __forceinline__ void async_wait_all() {
#if HAVE_ASYNC_LDS
    __builtin_amdgcn_s_wait_asynccnt(0);
#endif
}

// =====================================================================
// Prep 1: split W = W_hi(bf16) + W_lo(bf16 residual), stored TRANSPOSED
// as whiT/wloT[n][k] (k<256 -> W_in row k, else W_rec row k-256).
// =====================================================================
__global__ __launch_bounds__(256) void wsplit_prep_kernel(
    const float* __restrict__ W_in,    // [256, 2048]
    const float* __restrict__ W_rec,   // [2048, 2048]
    unsigned short* __restrict__ whiT, // [2048, 2304]
    unsigned short* __restrict__ wloT) // [2048, 2304]
{
    __shared__ unsigned int lds[64 * 65];
    const int k0 = blockIdx.x * 64;    // 36 blocks
    const int n0 = blockIdx.y * 64;    // 32 blocks
    const int tid = threadIdx.x;

    #pragma unroll
    for (int i = 0; i < 16; ++i) {
        int idx = tid + i * 256;
        int kk = idx >> 6, nn = idx & 63;
        int k = k0 + kk;
        float w = (k < N_INP) ? W_in[(long)k * N_RECC + n0 + nn]
                              : W_rec[(long)(k - N_INP) * N_RECC + n0 + nn];
        unsigned short hi = f32_to_bf16_rne(w);
        unsigned short lo = f32_to_bf16_rne(w - bf16_to_f32(hi));
        lds[kk * 65 + nn] = (unsigned int)hi | ((unsigned int)lo << 16);
    }
    __syncthreads();
    #pragma unroll
    for (int i = 0; i < 16; ++i) {
        int idx = tid + i * 256;
        int nn = idx >> 6, kk = idx & 63;
        unsigned int v = lds[kk * 65 + nn];
        long o = (long)(n0 + nn) * KTOT + k0 + kk;
        whiT[o] = (unsigned short)(v & 0xffffu);
        wloT[o] = (unsigned short)(v >> 16);
    }
}

// =====================================================================
// Prep 2: x (exact {0,1}) -> bf16 copy, same [B,T,N_INP] layout.
// =====================================================================
__global__ __launch_bounds__(256) void xconv_kernel(
    const float* __restrict__ x, unsigned short* __restrict__ xbf)
{
    long i = (long)blockIdx.x * 256 + threadIdx.x;   // 8 floats each
    const float4* src = (const float4*)x + i * 2;
    float4 a = src[0], b = src[1];
    uint4 u;
    u.x = pack_bf16(a.x, a.y);
    u.y = pack_bf16(a.z, a.w);
    u.z = pack_bf16(b.x, b.y);
    u.w = pack_bf16(b.z, b.w);
    *(uint4*)(xbf + i * 8) = u;
}

// =====================================================================
// Stage one K-chunk (A: 64x128, Bhi/Blo: 16x128 bf16) via 12 async b128
// copies per thread. Caller handles waits/barriers.
// =====================================================================
__device__ __forceinline__ void stage_chunk(
    int k0, int t, int tid, int n0,
    const unsigned short* __restrict__ xbf,
    const unsigned short* __restrict__ zrd,
    const unsigned short* __restrict__ whiT,
    const unsigned short* __restrict__ wloT,
    unsigned short* ldsA, unsigned short* ldsBhi, unsigned short* ldsBlo)
{
    const unsigned short* Asrc;
    long Astride;
    if (k0 < N_INP) { Asrc = xbf + (long)t * N_INP + k0; Astride = (long)TSTEPS * N_INP; }
    else            { Asrc = zrd + (k0 - N_INP);         Astride = N_RECC; }
    #pragma unroll
    for (int i = 0; i < 8; ++i) {
        int idx = tid + i * 128;
        int row = idx >> 4, c = idx & 15;
        async_cp16(Asrc + (long)row * Astride + c * 8, &ldsA[row * STR_BF + c * 8]);
    }
    #pragma unroll
    for (int i = 0; i < 2; ++i) {
        int idx = tid + i * 128;
        int nn = idx >> 4, c = idx & 15;
        async_cp16(whiT + (long)(n0 + nn) * KTOT + k0 + c * 8, &ldsBhi[nn * STR_BF + c * 8]);
    }
    #pragma unroll
    for (int i = 0; i < 2; ++i) {
        int idx = tid + i * 128;
        int nn = idx >> 4, c = idx & 15;
        async_cp16(wloT + (long)(n0 + nn) * KTOT + k0 + c * 8, &ldsBlo[nn * STR_BF + c * 8]);
    }
}

// =====================================================================
// bf16 split-precision LIF step (main path):
//   acc = A @ W_hi + A @ W_lo  via v_wmma_f32_16x16x32_bf16.
// LDS double-buffered: chunk i+1's async DMA overlaps chunk i's WMMAs
// (s_wait_asynccnt 12 = "previous chunk landed", in-order completion).
// zrd/zwr: parity-double-buffered bf16 spike state (avoids intra-step race).
// =====================================================================
__global__ __launch_bounds__(128) void lif_step_bf16_kernel(
    const unsigned short* __restrict__ xbf,   // [B, T, N_INP] bf16
    const float* __restrict__ noise,          // [B, T, N_RECC]
    const unsigned short* __restrict__ whiT,  // [2048, 2304]
    const unsigned short* __restrict__ wloT,  // [2048, 2304]
    const unsigned short* __restrict__ zrd,   // [B, N_RECC] bf16 z[t-1]
    unsigned short* __restrict__ zwr,         // [B, N_RECC] bf16 z[t]
    float* __restrict__ volt,                 // [B, T, N_RECC]
    float* __restrict__ spk,                  // [B, T, N_RECC]
    int t)
{
    __shared__ unsigned short ldsA[2][64 * STR_BF];
    __shared__ unsigned short ldsBhi[2][16 * STR_BF];
    __shared__ unsigned short ldsBlo[2][16 * STR_BF];

    const int tid  = threadIdx.x;
    const int lane = tid & 31;
    const int wave = tid >> 5;      // 0..3 -> batch rows [16w, 16w+16)
    const int half = lane >> 4;
    const int lr   = lane & 15;
    const int n0   = blockIdx.x * 16;

    v8f acc = {};
    const int Klim    = (t > 0) ? KTOT : N_INP;   // z[-1]==0: skip recurrent panel
    const int nchunks = Klim / KTILE_BF;          // 18 or 2

    // prologue: chunk 0 -> buffer 0
    stage_chunk(0, t, tid, n0, xbf, zrd, whiT, wloT, ldsA[0], ldsBhi[0], ldsBlo[0]);

    for (int ci = 0; ci < nchunks; ++ci) {
        const int cur = ci & 1;
        if (ci + 1 < nchunks) {
            // overlap: next chunk's DMA runs during this chunk's WMMAs
            stage_chunk((ci + 1) * KTILE_BF, t, tid, n0, xbf, zrd, whiT, wloT,
                        ldsA[cur ^ 1], ldsBhi[cur ^ 1], ldsBlo[cur ^ 1]);
            __builtin_prefetch(whiT + (long)n0 * KTOT + (ci + 1) * KTILE_BF, 0, 1);
            async_wait_prev();    // s_wait_asynccnt 12 -> chunk ci landed (ours)
        } else {
            async_wait_all();     // s_wait_asynccnt 0
        }
        __syncthreads();          // chunk ci landed for ALL waves

        // ---- 4 K-steps x (hi + lo) = 8 x v_wmma_f32_16x16x32_bf16 ----
        // 16-bit A 16x32 layout: lane holds K = [8*half, +8) and [16+8*half, +8)
        const unsigned short* pa = &ldsA  [cur][(wave * 16 + lr) * STR_BF + 8 * half];
        const unsigned short* ph = &ldsBhi[cur][lr * STR_BF + 8 * half];
        const unsigned short* pl = &ldsBlo[cur][lr * STR_BF + 8 * half];
        #pragma unroll
        for (int kk = 0; kk < KTILE_BF; kk += 32) {
            v8bf a0 = *(const v8bf*)&pa[kk];
            v8bf a1 = *(const v8bf*)&pa[kk + 16];
            v16bf a = __builtin_shufflevector(a0, a1, 0,1,2,3,4,5,6,7,8,9,10,11,12,13,14,15);
            v8bf h0 = *(const v8bf*)&ph[kk];
            v8bf h1 = *(const v8bf*)&ph[kk + 16];
            v16bf bh = __builtin_shufflevector(h0, h1, 0,1,2,3,4,5,6,7,8,9,10,11,12,13,14,15);
            v8bf l0 = *(const v8bf*)&pl[kk];
            v8bf l1 = *(const v8bf*)&pl[kk + 16];
            v16bf bl = __builtin_shufflevector(l0, l1, 0,1,2,3,4,5,6,7,8,9,10,11,12,13,14,15);
            acc = __builtin_amdgcn_wmma_f32_16x16x32_bf16(false, a, false, bh, (short)0, acc, false, false);
            acc = __builtin_amdgcn_wmma_f32_16x16x32_bf16(false, a, false, bl, (short)0, acc, false, false);
        }
        __syncthreads();          // all readers done before buf[cur] is re-staged at ci+2
    }

    // ---- epilogue: LIF update + threshold ----
    const int n = n0 + lr;
    #pragma unroll
    for (int r = 0; r < 8; ++r) {
        const int  b     = wave * 16 + r + half * 8;
        const long o_cur = ((long)b * TSTEPS + t) * N_RECC + n;
        float vprev = 0.f, zprev = 0.f;
        if (t > 0) {
            vprev = volt[o_cur - N_RECC];
            zprev = spk [o_cur - N_RECC];
        }
        const float i_t  = acc[r] + noise[o_cur];
        const float vnew = DECAYF * vprev + i_t - zprev * THRF;
        const bool  fire = (vnew > THRF);
        volt[o_cur] = vnew;
        spk [o_cur] = fire ? 1.0f : 0.0f;
        zwr[(long)b * N_RECC + n] = fire ? (unsigned short)0x3F80 : (unsigned short)0; // bf16 1.0/0.0
    }
}

// =====================================================================
// f32 WMMA fallback path (bit-faithful, used when ws_size is too small)
// =====================================================================
__device__ __forceinline__ void gemm_panel_f32(const float* __restrict__ Asrc, long Astride,
                                               const float* __restrict__ Bsrc, int K,
                                               float* ldsA, float* ldsB,
                                               int tid, int wave, int lr, int half,
                                               v8f& acc)
{
    for (int k0 = 0; k0 < K; k0 += KTILE_F32) {
        __syncthreads();
        #pragma unroll
        for (int i = 0; i < 8; ++i) {
            int idx = tid + i * 128;
            int row = idx >> 4, q = idx & 15;
            float4 v = *(const float4*)(Asrc + (long)row * Astride + k0 + q * 4);
            *(float4*)&ldsA[row * STR_F32 + q * 4] = v;
        }
        #pragma unroll
        for (int i = 0; i < 2; ++i) {
            int idx = tid + i * 128;
            int k = idx >> 2, c4 = idx & 3;
            float4 v = *(const float4*)(Bsrc + (long)(k0 + k) * N_RECC + c4 * 4);
            ldsB[(c4 * 4 + 0) * STR_F32 + k] = v.x;
            ldsB[(c4 * 4 + 1) * STR_F32 + k] = v.y;
            ldsB[(c4 * 4 + 2) * STR_F32 + k] = v.z;
            ldsB[(c4 * 4 + 3) * STR_F32 + k] = v.w;
        }
        __syncthreads();
        const float* pa = &ldsA[(wave * 16 + lr) * STR_F32];
        const float* pb = &ldsB[lr * STR_F32];
        #pragma unroll
        for (int kk = 0; kk < KTILE_F32; kk += 4) {
            v2f a = *(const v2f*)&pa[kk + 2 * half];
            v2f b = *(const v2f*)&pb[kk + 2 * half];
            acc = __builtin_amdgcn_wmma_f32_16x16x4_f32(false, a, false, b, (short)0, acc, false, false);
        }
    }
}

__global__ __launch_bounds__(128) void lif_step_f32_kernel(
    const float* __restrict__ x, const float* __restrict__ noise,
    const float* __restrict__ W_in, const float* __restrict__ W_rec,
    float* __restrict__ volt, float* __restrict__ spk, int t)
{
    __shared__ float ldsA[64 * STR_F32];
    __shared__ float ldsB[16 * STR_F32];

    const int tid  = threadIdx.x;
    const int lane = tid & 31;
    const int wave = tid >> 5;
    const int half = lane >> 4;
    const int lr   = lane & 15;
    const int n0   = blockIdx.x * 16;

    v8f acc = {};
    gemm_panel_f32(x + (long)t * N_INP, (long)TSTEPS * N_INP,
                   W_in + n0, N_INP, ldsA, ldsB, tid, wave, lr, half, acc);
    if (t > 0) {
        gemm_panel_f32(spk + (long)(t - 1) * N_RECC, (long)TSTEPS * N_RECC,
                       W_rec + n0, N_RECC, ldsA, ldsB, tid, wave, lr, half, acc);
    }

    const int n = n0 + lr;
    #pragma unroll
    for (int r = 0; r < 8; ++r) {
        const int  b     = wave * 16 + r + half * 8;
        const long o_cur = ((long)b * TSTEPS + t) * N_RECC + n;
        float vprev = 0.f, zprev = 0.f;
        if (t > 0) { vprev = volt[o_cur - N_RECC]; zprev = spk[o_cur - N_RECC]; }
        const float i_t  = acc[r] + noise[o_cur];
        const float vnew = DECAYF * vprev + i_t - zprev * THRF;
        volt[o_cur] = vnew;
        spk [o_cur] = (vnew > THRF) ? 1.0f : 0.0f;
    }
}

// =====================================================================
extern "C" void kernel_launch(void* const* d_in, const int* in_sizes, int n_in,
                              void* d_out, int out_size, void* d_ws, size_t ws_size,
                              hipStream_t stream)
{
    (void)in_sizes; (void)n_in; (void)out_size;

    const float* x     = (const float*)d_in[0];
    const float* noise = (const float*)d_in[1];
    const float* W_in  = (const float*)d_in[2];
    const float* W_rec = (const float*)d_in[3];

    float* volt = (float*)d_out;
    float* spk  = (float*)d_out + (size_t)BATCH * TSTEPS * N_RECC;

    // ws layout (ushort units): whiT | wloT | xbf | zbuf[2]
    const size_t WSPLIT = (size_t)N_RECC * KTOT;            // 4,718,592
    const size_t XELEMS = (size_t)BATCH * TSTEPS * N_INP;   // 8,192,000
    const size_t ZELEMS = (size_t)BATCH * N_RECC;           // 131,072
    const size_t need = (2 * WSPLIT + XELEMS + 2 * ZELEMS) * sizeof(unsigned short); // ~34 MiB

    dim3 grid(N_RECC / 16);   // 128 WGs, one 16-col tile each
    dim3 block(128);          // 4 wave32s, one 16x16 WMMA tile per wave

    if (ws_size >= need) {
        unsigned short* whiT = (unsigned short*)d_ws;
        unsigned short* wloT = whiT + WSPLIT;
        unsigned short* xbf  = wloT + WSPLIT;
        unsigned short* zbuf = xbf + XELEMS;

        dim3 pg(KTOT / 64, N_RECC / 64), pb(256);
        hipLaunchKernelGGL(wsplit_prep_kernel, pg, pb, 0, stream, W_in, W_rec, whiT, wloT);
        hipLaunchKernelGGL(xconv_kernel, dim3((unsigned)(XELEMS / 8 / 256)), dim3(256), 0, stream,
                           x, xbf);
        for (int t = 0; t < TSTEPS; ++t) {
            const unsigned short* zrd = zbuf + (size_t)(t & 1) * ZELEMS;
            unsigned short*       zwr = zbuf + (size_t)((t + 1) & 1) * ZELEMS;
            hipLaunchKernelGGL(lif_step_bf16_kernel, grid, block, 0, stream,
                               xbf, noise, whiT, wloT, zrd, zwr, volt, spk, t);
        }
    } else {
        for (int t = 0; t < TSTEPS; ++t) {
            hipLaunchKernelGGL(lif_step_f32_kernel, grid, block, 0, stream,
                               x, noise, W_in, W_rec, volt, spk, t);
        }
    }
}
